// ContrastiveLoss3_47528108098354
// MI455X (gfx1250) — matrix operand order repaired and checked by reading, not compiled, not run on hardware
//
#include <hip/hip_runtime.h>
#include <hip/hip_bf16.h>

// Problem constants (from reference): B=4096 rows per view, D=128, TEMP=0.25
#define BB 4096
#define DD 128

// (1/TEMP) * log2(e): sim/T in exp() done as exp2(sim * kExpScale) on v_exp_f32
__device__ constexpr float kExpScale = 5.7707801635558544f;
__device__ constexpr float kLn2      = 0.6931471805599453f;

typedef __attribute__((ext_vector_type(16))) _Float16 v16h;
typedef __attribute__((ext_vector_type(8)))  _Float16 v8h;
typedef __attribute__((ext_vector_type(4)))  _Float16 v4h;
typedef __attribute__((ext_vector_type(8)))  float    v8f;
typedef __attribute__((ext_vector_type(4)))  float    v4f;

// ---------------------------------------------------------------------------
// Kernel 1: row-normalize both views (fp32), emit f16 copies for WMMA and the
// exact fp32 positive dot products diag[k] = z_i[k] . z_j[k].
// One wave (32 lanes) per row; 4 floats per lane.
// ---------------------------------------------------------------------------
__global__ __launch_bounds__(256) void norm_diag_kernel(
    const float* __restrict__ ei, const float* __restrict__ ej,
    _Float16* __restrict__ zi, _Float16* __restrict__ zj,
    float* __restrict__ diag) {
  const int row  = (blockIdx.x * blockDim.x + threadIdx.x) >> 5;
  const int lane = threadIdx.x & 31;

  const v4f a = ((const v4f*)(ei + (size_t)row * DD))[lane];
  const v4f b = ((const v4f*)(ej + (size_t)row * DD))[lane];

  float si = a[0]*a[0] + a[1]*a[1] + a[2]*a[2] + a[3]*a[3];
  float sj = b[0]*b[0] + b[1]*b[1] + b[2]*b[2] + b[3]*b[3];
  float dd = a[0]*b[0] + a[1]*b[1] + a[2]*b[2] + a[3]*b[3];
#pragma unroll
  for (int off = 1; off < 32; off <<= 1) {
    si += __shfl_xor(si, off, 32);
    sj += __shfl_xor(sj, off, 32);
    dd += __shfl_xor(dd, off, 32);
  }
  const float ri = 1.0f / fmaxf(sqrtf(si), 1e-12f);
  const float rj = 1.0f / fmaxf(sqrtf(sj), 1e-12f);

  v4h ha, hb;
#pragma unroll
  for (int c = 0; c < 4; ++c) {
    ha[c] = (_Float16)(a[c] * ri);
    hb[c] = (_Float16)(b[c] * rj);
  }
  ((v4h*)(zi + (size_t)row * DD))[lane] = ha;
  ((v4h*)(zj + (size_t)row * DD))[lane] = hb;
  if (lane == 0) diag[row] = dd * ri * rj;
}

// ---------------------------------------------------------------------------
// Kernel 2: fused  out[m] = sum_n exp( (A[m,:] . Bm[n,:]) / T )   (m,n in [0,4096))
// A, Bm are row-major f16 [4096 x 128] unit vectors.
// Each wave owns 32 rows (two 16-row WMMA groups); A fragments live in
// registers across the whole column sweep. Column loop: 16-wide tiles,
// K=128 via 4x v_wmma_f32_16x16x32_f16, exp via v_exp_f32, row sums reduced
// with xor shuffles within each 16-lane half. Deterministic, no atomics.
//
// ISA fragment layouts (CDNA5 ISA §7.12.2), wave32:
//   A 16x32 f16:  lane<16 -> row=lane,  elems0..7 = K{0..7},  elems8..15 = K{16..23}
//                 lane>=16 -> row=lane-16, K{8..15} then K{24..31}
//   B 32x16 f16:  lane<16 -> col=lane,  elems0..15 = K{0..15} (this code's order, A too; ISA: same as A)
//                 lane>=16 -> col=lane-16, K{16..31}
//   C/D f32:      c[v] = S[ 8*half + v ][ lane&15 ]   per 16x16 tile
// ---------------------------------------------------------------------------
__global__ __launch_bounds__(128) void gemm_exp_rowsum_kernel(
    const _Float16* __restrict__ A, const _Float16* __restrict__ Bm,
    float* __restrict__ out) {
  const int tid  = threadIdx.x;
  const int wave = blockIdx.x * (blockDim.x >> 5) + (tid >> 5);
  const int lane = tid & 31;
  const int half = lane >> 4;   // 0: lanes 0-15, 1: lanes 16-31
  const int mrow = lane & 15;
  const int m0   = wave * 32;   // this wave's first output row

  // Preload A fragments: 2 row-groups x 4 K-steps (held in VGPRs all pass)
  v16h afrag[2][4];
#pragma unroll
  for (int g = 0; g < 2; ++g) {
    const _Float16* rp = A + (size_t)(m0 + 16 * g + mrow) * DD;
#pragma unroll
    for (int k = 0; k < 4; ++k) {
      const int k0 = 32 * k;
      const v16h a = *(const v16h*)(rp + k0 + 16 * half);      // K: k0+16h .. +15
      afrag[g][k] = a;
    }
  }

  v8f acc0 = {};  // rows m0      + 8*half + v
  v8f acc1 = {};  // rows m0 + 16 + 8*half + v

  for (int n0 = 0; n0 < BB; n0 += 16) {
    // This lane's B column = row (n0 + mrow) of Bm; 32B contiguous per K-step.
    const _Float16* cp = Bm + (size_t)(n0 + mrow) * DD + 16 * half;
    __builtin_prefetch(cp + 16 * DD, 0, 0);  // next tile's column row

    v8f c0 = {};
    v8f c1 = {};
#pragma unroll
    for (int k = 0; k < 4; ++k) {
      const v16h b = *(const v16h*)(cp + 32 * k);
      c0 = __builtin_amdgcn_wmma_f32_16x16x32_f16(false, afrag[0][k], false, b,
                                                  (short)0, c0, false, false);
      c1 = __builtin_amdgcn_wmma_f32_16x16x32_f16(false, afrag[1][k], false, b,
                                                  (short)0, c1, false, false);
    }
#pragma unroll
    for (int v = 0; v < 8; ++v) {
      acc0[v] += __builtin_amdgcn_exp2f(c0[v] * kExpScale);
      acc1[v] += __builtin_amdgcn_exp2f(c1[v] * kExpScale);
    }
  }

  // Reduce across the 16 lanes of each half (bit4 of lane untouched).
#pragma unroll
  for (int off = 1; off < 16; off <<= 1) {
#pragma unroll
    for (int v = 0; v < 8; ++v) {
      acc0[v] += __shfl_xor(acc0[v], off, 32);
      acc1[v] += __shfl_xor(acc1[v], off, 32);
    }
  }

  if (mrow == 0) {
    float* o0 = out + m0 + 8 * half;
    float* o1 = out + m0 + 16 + 8 * half;
#pragma unroll
    for (int v = 0; v < 8; ++v) { o0[v] = acc0[v]; o1[v] = acc1[v]; }
  }
}

// ---------------------------------------------------------------------------
// Kernel 3: loss = (1/2B) * sum_k [ -8*diag[k] + ln(rowsum[k]) + ln(colsum[k]) ]
// ---------------------------------------------------------------------------
__global__ __launch_bounds__(256) void loss_reduce_kernel(
    const float* __restrict__ rowsum, const float* __restrict__ colsum,
    const float* __restrict__ diag, float* __restrict__ out) {
  float s = 0.0f;
  for (int k = threadIdx.x; k < BB; k += 256) {
    s += -8.0f * diag[k] +
         kLn2 * (__builtin_amdgcn_logf(rowsum[k]) +
                 __builtin_amdgcn_logf(colsum[k]));
  }
  __shared__ float red[256];
  red[threadIdx.x] = s;
  __syncthreads();
#pragma unroll
  for (int st = 128; st > 0; st >>= 1) {
    if (threadIdx.x < st) red[threadIdx.x] += red[threadIdx.x + st];
    __syncthreads();
  }
  if (threadIdx.x == 0) out[0] = red[0] / (2.0f * BB);
}

// ---------------------------------------------------------------------------
extern "C" void kernel_launch(void* const* d_in, const int* in_sizes, int n_in,
                              void* d_out, int out_size, void* d_ws, size_t ws_size,
                              hipStream_t stream) {
  (void)in_sizes; (void)n_in; (void)out_size; (void)ws_size;

  const float* ei = (const float*)d_in[0];  // emb_i [4096,128] f32
  const float* ej = (const float*)d_in[1];  // emb_j [4096,128] f32
  float* out = (float*)d_out;               // scalar loss

  char* ws = (char*)d_ws;
  _Float16* zi     = (_Float16*)ws;                                  // 1 MB
  _Float16* zj     = (_Float16*)(ws + (size_t)BB * DD * 2);          // 1 MB
  float*    rowsum = (float*)(ws + (size_t)2 * BB * DD * 2);         // 16 KB
  float*    colsum = rowsum + BB;                                    // 16 KB
  float*    diag   = colsum + BB;                                    // 16 KB

  // 1) Normalize + exact fp32 positives. 1 wave/row, 8 waves/block.
  norm_diag_kernel<<<BB / 8, 256, 0, stream>>>(ei, ej, zi, zj, diag);

  // 2) rowsum = row sums of exp(Zi.Zj^T / T); colsum = row sums of the
  //    transposed product (== column sums). 32 rows/wave, 4 waves/block.
  gemm_exp_rowsum_kernel<<<BB / 128, 128, 0, stream>>>(zi, zj, rowsum);
  gemm_exp_rowsum_kernel<<<BB / 128, 128, 0, stream>>>(zj, zi, colsum);

  // 3) Final scalar reduction.
  loss_reduce_kernel<<<1, 256, 0, stream>>>(rowsum, colsum, diag, out);
}
